// TtMistralAttention_17514876634000
// MI455X (gfx1250) — matrix operand order repaired and checked by reading, not compiled, not run on hardware
//
#include <hip/hip_runtime.h>
#include <hip/hip_bf16.h>
#include <math.h>

// ---------------------------------------------------------------------------
// Mistral decode attention for MI455X (gfx1250, wave32, WMMA).
// Memory-bound workload (~1.2 GB traffic, ~50us floor @ 23.3 TB/s):
//  - flash-decode streaming attention (1024 WGs) with bf16 WMMA
//  - split-K WMMA GEMMs for the projections (192/128 WGs)
//  - async global->LDS staging (GLOBAL_LOAD_ASYNC_TO_LDS_B128 + s_wait_asynccnt)
//    with double buffering to overlap weight streaming with WMMA.
// ---------------------------------------------------------------------------

#define DIM    4096
#define NH     32
#define NKV    8
#define HD     128
#define BATCH  32
#define KVLEN  4096
#define WINDOW 4096
#define G_     4          // NH / NKV
#define NSPLIT 4          // attention length splits
#define NPART  16         // NSPLIT * 4 waves
#define PSTRIDE (4*HD + 8)
#define NSPLITK 4         // GEMM split-K factor

typedef __bf16 bf16_t;
typedef __attribute__((ext_vector_type(16))) __bf16 v16bf;
typedef __attribute__((ext_vector_type(8)))  float  v8f;

static __device__ __forceinline__ int imin(int a, int b) { return a < b ? a : b; }

// Element -> K index mapping for the 16-bit A-matrix 16x32 fragment
// (contiguous, matching B): lanes 0-15 hold K={0..15}, lanes 16-31 hold K={16..31}.
static __device__ __forceinline__ int a_frag_k(int j, int half) {
  return half * 16 + j;
}

static __device__ __forceinline__ v8f wmma_bf16(v16bf a, v16bf b, v8f c) {
  return __builtin_amdgcn_wmma_f32_16x16x32_bf16(false, a, false, b, (short)0, c,
                                                 false, false);
}

// ---- CDNA5 async global->LDS copy (ASYNCcnt path, ISA ch.10 / 08_async) ----
static __device__ __forceinline__ void async_copy_b128(void* lds_dst,
                                                       const void* gaddr) {
  unsigned lds_off = (unsigned)(uintptr_t)lds_dst;  // low 32b = LDS byte offset
  asm volatile("global_load_async_to_lds_b128 %0, %1, off"
               :: "v"(lds_off), "v"(gaddr) : "memory");
}
static __device__ __forceinline__ void wait_async_0() {
  asm volatile("s_wait_asynccnt 0" ::: "memory");
}
static __device__ __forceinline__ void wait_async_4() {
  asm volatile("s_wait_asynccnt 4" ::: "memory");
}

// ---------------------------------------------------------------------------
// Split-K GEMM: Cpart[split] (32xN) = A(32xK[split]) @ B(K[split]xN).
// Block: 256 threads = 8 waves, each wave owns a 16-col tile (block = 128 cols).
// B tiles stream via double-buffered async LDS copies.
// ---------------------------------------------------------------------------
__global__ void wmma_gemm32_splitk(const float* __restrict__ A,
                                   const float* __restrict__ B,
                                   float* __restrict__ Cpart, int K, int N) {
  __shared__ float As[32 * 32];
  __shared__ float Bs[2][32 * 128];
  const int tid  = threadIdx.x;
  const int wave = tid >> 5;
  const int lane = tid & 31;
  const int half = lane >> 4;
  const int nloc = lane & 15;
  const int colb = blockIdx.x * 128;
  const int col0 = colb + wave * 16;
  const int split  = blockIdx.y;
  const int kchunk = K / NSPLITK;
  const int kbeg   = split * kchunk;
  const int nt     = kchunk / 32;

  v8f acc0 = {}; v8f acc1 = {};

  { // issue tile 0 -> buf 0 (4 async b128 per thread)
#pragma unroll
    for (int j = 0; j < 4; ++j) {
      int idx = j * 256 + tid;          // float4 index
      int kr = idx >> 5, cs = idx & 31;
      async_copy_b128(&Bs[0][kr * 128 + cs * 4],
                      &B[(size_t)(kbeg + kr) * N + colb + cs * 4]);
    }
  }

  for (int t = 0; t < nt; ++t) {
    const int k0  = kbeg + t * 32;
    const int cur = t & 1;
    __syncthreads();                    // all prior reads of both buffers done
    if (t + 1 < nt) {                   // issue tile t+1 -> other buffer
#pragma unroll
      for (int j = 0; j < 4; ++j) {
        int idx = j * 256 + tid;
        int kr = idx >> 5, cs = idx & 31;
        async_copy_b128(&Bs[1 - cur][kr * 128 + cs * 4],
                        &B[(size_t)(k0 + 32 + kr) * N + colb + cs * 4]);
      }
    }
    { // stage A tile 32x32 synchronously (small)
      int idx = tid * 4;
      int r = idx >> 5, c = idx & 31;
      *(float4*)&As[idx] = *(const float4*)&A[r * K + k0 + c];
    }
    if (t + 1 < nt) wait_async_4();     // current tile landed (in-order)
    else            wait_async_0();
    __syncthreads();

    v16bf a0, a1, bf;
#pragma unroll
    for (int j = 0; j < 16; ++j) {
      int kk = a_frag_k(j, half);
      a0[j] = (bf16_t)As[nloc * 32 + kk];
      a1[j] = (bf16_t)As[(16 + nloc) * 32 + kk];
      bf[j] = (bf16_t)Bs[cur][(half * 16 + j) * 128 + wave * 16 + nloc];
    }
    acc0 = wmma_bf16(a0, bf, acc0);
    acc1 = wmma_bf16(a1, bf, acc1);
  }

  float* C = &Cpart[(size_t)split * 32 * N];
#pragma unroll
  for (int r = 0; r < 8; ++r) {
    int m0 = r + half * 8;
    C[m0 * N + col0 + nloc]        = acc0[r];
    C[(16 + m0) * N + col0 + nloc] = acc1[r];
  }
}

__global__ void reduce_splitk(const float* __restrict__ Cpart,
                              float* __restrict__ C, int total) {
  int i = blockIdx.x * 256 + threadIdx.x;
  if (i < total) {
    float s = 0.0f;
#pragma unroll
    for (int p = 0; p < NSPLITK; ++p) s += Cpart[(size_t)p * total + i];
    C[i] = s;
  }
}

// ---------------------------------------------------------------------------
// RoPE: rows @ rot_mat(128x128).  Blocks 0..63: Q (1024 rows); 64..79: K
// (256 rows) + V copy.  16 rows per block, 8 waves -> 8 col tiles.
// rot_mat tiles stream via async LDS copies.
// ---------------------------------------------------------------------------
__global__ void rope_kernel(const float* __restrict__ xqkv, const float* __restrict__ rot,
                            float* __restrict__ q_rot, float* __restrict__ k_rot,
                            float* __restrict__ v_new) {
  __shared__ float As[16 * 32];
  __shared__ float Bs[32 * 128];
  const bool is_k  = blockIdx.x >= 64;
  const int rowblk = is_k ? (blockIdx.x - 64) : blockIdx.x;
  const int tid  = threadIdx.x;
  const int wave = tid >> 5;
  const int lane = tid & 31;
  const int half = lane >> 4;
  const int nloc = lane & 15;

  v8f acc = {};
  for (int k0 = 0; k0 < HD; k0 += 32) {
#pragma unroll
    for (int j = 0; j < 4; ++j) { // async stage rot rows k0..k0+31
      int idx = j * 256 + tid;
      int kr = idx >> 5, cs = idx & 31;
      async_copy_b128(&Bs[kr * 128 + cs * 4], &rot[(k0 + kr) * HD + cs * 4]);
    }
    { // stage A 16x32 (512 floats, float2 per thread)
      int idx = tid * 2;
      int r = idx >> 5, c = idx & 31;
      int row = rowblk * 16 + r;
      const float* src = is_k
          ? &xqkv[(row >> 3) * 6144 + 4096 + (row & 7) * HD]
          : &xqkv[(row >> 5) * 6144 + (row & 31) * HD];
      *(float2*)&As[idx] = *(const float2*)&src[k0 + c];
    }
    wait_async_0();
    __syncthreads();

    v16bf a, bf;
#pragma unroll
    for (int j = 0; j < 16; ++j) {
      int kk = a_frag_k(j, half);
      a[j]  = (bf16_t)As[nloc * 32 + kk];
      bf[j] = (bf16_t)Bs[(half * 16 + j) * 128 + wave * 16 + nloc];
    }
    acc = wmma_bf16(a, bf, acc);
    __syncthreads();
  }

  float* dst = is_k ? k_rot : q_rot;
#pragma unroll
  for (int r = 0; r < 8; ++r) {
    int m = r + half * 8;
    dst[(rowblk * 16 + m) * HD + wave * 16 + nloc] = acc[r];
  }
  if (is_k) { // copy new V rows out of xqkv
#pragma unroll
    for (int j = 0; j < 8; ++j) {
      int idx = j * 256 + tid;
      int r = idx >> 7, d = idx & 127;
      int row = rowblk * 16 + r;
      v_new[row * HD + d] = xqkv[(row >> 3) * 6144 + 5120 + (row & 7) * HD + d];
    }
  }
}

// ---------------------------------------------------------------------------
// Flash-decode attention. Grid = B*NKV*NSPLIT, 128 threads (4 waves).
// Each wave streams 32 keys per iteration: 8 WMMAs for scores (Q x K^T over
// 128 dims) + online softmax + 8 WMMAs for P x V.  Key l == pos substitutes
// the freshly-rotated K/V (cache inputs are never mutated).
// ---------------------------------------------------------------------------
__global__ void attn_kernel(const float* __restrict__ q_rot, const float* __restrict__ k_rot,
                            const float* __restrict__ v_new,
                            const float* __restrict__ cache_k, const float* __restrict__ cache_v,
                            const float* __restrict__ mask, const int* __restrict__ pos_p,
                            float* __restrict__ partials) {
  const int pos  = pos_p[0];
  int Ltot = ((pos + 1 + 31) >> 5) << 5;
  if (Ltot > WINDOW) Ltot = WINDOW;

  const int blk = blockIdx.x;
  const int s   = blk & (NSPLIT - 1);
  const int kv  = (blk / NSPLIT) & (NKV - 1);
  const int b   = blk / (NSPLIT * NKV);
  const int tid  = threadIdx.x;
  const int wave = tid >> 5;
  const int lane = tid & 31;
  const int half = lane >> 4;
  const int nloc = lane & 15;
  const int bk   = b * NKV + kv;

  __shared__ float Qs[G_ * HD];
  __shared__ float Ps[4][16 * 32];

  { // stage Q (4 heads x 128) into LDS
    int idx = tid * 4;
    int r = idx >> 7, d = idx & 127;
    *(float4*)&Qs[idx] = *(const float4*)&q_rot[(b * NH + kv * G_ + r) * HD + d];
  }
  __syncthreads();

  // Q fragments (M=16 rows: 4 valid heads + 12 zero pad), one per 32-dim step.
  v16bf aq[4];
#pragma unroll
  for (int st = 0; st < 4; ++st)
#pragma unroll
    for (int j = 0; j < 16; ++j) {
      int kk = st * 32 + a_frag_k(j, half);
      aq[st][j] = (nloc < G_) ? (bf16_t)Qs[nloc * HD + kk] : (bf16_t)0.0f;
    }

  float m_state[8], l_state[8];
#pragma unroll
  for (int r = 0; r < 8; ++r) { m_state[r] = -3.0e38f; l_state[r] = 0.0f; }
  v8f outacc[8];
#pragma unroll
  for (int t = 0; t < 8; ++t) outacc[t] = (v8f){};

  const float scale = 0.08838834764831845f; // 1/sqrt(128)
  const int Ls = Ltot / NSPLIT;
  const float* krow_new = &k_rot[bk * HD];
  const float* vrow_new = &v_new[bk * HD];
  const float* kbase = &cache_k[(size_t)bk * KVLEN * HD];
  const float* vbase = &cache_v[(size_t)bk * KVLEN * HD];

  for (int kb0 = s * Ls; kb0 < s * Ls + Ls; kb0 += 128) {
    const int kb = kb0 + wave * 32;
    // prefetch next chunk of K (global_prefetch_b8)
    if (kb + 128 < Ltot) __builtin_prefetch(&kbase[(size_t)(kb + 128) * HD], 0, 1);

    // ---- scores: two 16(head)x16(key) tiles ----
    v8f sc[2];
    sc[0] = (v8f){}; sc[1] = (v8f){};
#pragma unroll
    for (int c = 0; c < 2; ++c) {
      int key  = kb + c * 16 + nloc;
      int keyc = imin(key, Ltot - 1);
      const float* kptr = (keyc == pos) ? krow_new : &kbase[(size_t)keyc * HD];
#pragma unroll
      for (int st = 0; st < 4; ++st) {
        v16bf bkf;
#pragma unroll
        for (int jj = 0; jj < 4; ++jj) {
          float4 f = *(const float4*)&kptr[st * 32 + half * 16 + jj * 4];
          bkf[jj * 4 + 0] = (bf16_t)f.x; bkf[jj * 4 + 1] = (bf16_t)f.y;
          bkf[jj * 4 + 2] = (bf16_t)f.z; bkf[jj * 4 + 3] = (bf16_t)f.w;
        }
        sc[c] = wmma_bf16(aq[st], bkf, sc[c]);
      }
    }

    // ---- scale + mask + validity ----
#pragma unroll
    for (int c = 0; c < 2; ++c) {
      int key = kb + c * 16 + nloc;
#pragma unroll
      for (int r = 0; r < 8; ++r) {
        float v = sc[c][r] * scale;
        int m = r + half * 8;
        if (m < G_ && key < Ltot)
          v += mask[(size_t)((kv * G_ + m) * BATCH + b) * WINDOW + key];
        if (key >= Ltot) v = -3.0e38f;
        sc[c][r] = v;
      }
    }

    // ---- online softmax (row = head, reduce across 16-lane column group) ----
    float alpha[8];
#pragma unroll
    for (int r = 0; r < 8; ++r) {
      float mx = fmaxf(sc[0][r], sc[1][r]);
      mx = fmaxf(mx, __shfl_xor(mx, 1, 32));
      mx = fmaxf(mx, __shfl_xor(mx, 2, 32));
      mx = fmaxf(mx, __shfl_xor(mx, 4, 32));
      mx = fmaxf(mx, __shfl_xor(mx, 8, 32));
      float mn = fmaxf(m_state[r], mx);
      alpha[r] = __expf(m_state[r] - mn);
      m_state[r] = mn;
      int key0 = kb + nloc, key1 = kb + 16 + nloc;
      float p0 = (key0 < Ltot) ? __expf(sc[0][r] - mn) : 0.0f;
      float p1 = (key1 < Ltot) ? __expf(sc[1][r] - mn) : 0.0f;
      sc[0][r] = p0; sc[1][r] = p1;
      float sm = p0 + p1;
      sm += __shfl_xor(sm, 1, 32);
      sm += __shfl_xor(sm, 2, 32);
      sm += __shfl_xor(sm, 4, 32);
      sm += __shfl_xor(sm, 8, 32);
      l_state[r] = l_state[r] * alpha[r] + sm;
    }
#pragma unroll
    for (int t = 0; t < 8; ++t)
#pragma unroll
      for (int r = 0; r < 8; ++r) outacc[t][r] *= alpha[r];

    // ---- re-layout P (C-layout -> A-layout) through LDS ----
#pragma unroll
    for (int c = 0; c < 2; ++c)
#pragma unroll
      for (int r = 0; r < 8; ++r)
        Ps[wave][(r + half * 8) * 32 + c * 16 + nloc] = sc[c][r];
    __syncthreads();

    v16bf pa;
#pragma unroll
    for (int j = 0; j < 16; ++j)
      pa[j] = (bf16_t)Ps[wave][nloc * 32 + a_frag_k(j, half)];

    // ---- P x V: 8 dim tiles ----
#pragma unroll
    for (int t = 0; t < 8; ++t) {
      v16bf bv;
#pragma unroll
      for (int j = 0; j < 16; ++j) {
        int key  = kb + half * 16 + j;
        int keyc = imin(key, Ltot - 1);
        const float* vptr = (keyc == pos) ? vrow_new : &vbase[(size_t)keyc * HD];
        bv[j] = (bf16_t)vptr[t * 16 + nloc];
      }
      outacc[t] = wmma_bf16(pa, bv, outacc[t]);
    }
    __syncthreads();
  }

  // ---- write per-wave partial: out(4x128), m[4], l[4] ----
  float* dst = &partials[(size_t)(bk * NPART + s * 4 + wave) * PSTRIDE];
  if (half == 0) {
#pragma unroll
    for (int r = 0; r < G_; ++r)
#pragma unroll
      for (int t = 0; t < 8; ++t)
        dst[r * HD + t * 16 + nloc] = outacc[t][r];
  }
  if (lane == 0) {
#pragma unroll
    for (int r = 0; r < G_; ++r) {
      dst[G_ * HD + r]      = m_state[r];
      dst[G_ * HD + 4 + r]  = l_state[r];
    }
  }
}

// ---------------------------------------------------------------------------
// Log-sum-exp combine of NPART partials per (b, kv); writes (B, NH*HD).
// ---------------------------------------------------------------------------
__global__ void combine_kernel(const float* __restrict__ partials,
                               float* __restrict__ attn_out) {
  const int bk = blockIdx.x;
  const int b  = bk >> 3;
  const int kv = bk & 7;
  const int d  = threadIdx.x;
  const float* base = &partials[(size_t)bk * NPART * PSTRIDE];
#pragma unroll
  for (int g = 0; g < G_; ++g) {
    float M = -3.0e38f;
    for (int p = 0; p < NPART; ++p)
      M = fmaxf(M, base[p * PSTRIDE + G_ * HD + g]);
    float denom = 0.0f, acc = 0.0f;
    for (int p = 0; p < NPART; ++p) {
      float w = __expf(base[p * PSTRIDE + G_ * HD + g] - M);
      denom += base[p * PSTRIDE + G_ * HD + 4 + g] * w;
      acc   += base[p * PSTRIDE + g * HD + d] * w;
    }
    attn_out[(size_t)b * DIM + (kv * G_ + g) * HD + d] = acc / denom;
  }
}

// ---------------------------------------------------------------------------
extern "C" void kernel_launch(void* const* d_in, const int* in_sizes, int n_in,
                              void* d_out, int out_size, void* d_ws, size_t ws_size,
                              hipStream_t stream) {
  const float* x        = (const float*)d_in[0];   // (1,1,32,4096)
  const float* wqkv     = (const float*)d_in[1];   // (4096,6144)
  const float* wo       = (const float*)d_in[2];   // (4096,4096)
  const float* rot_mat  = (const float*)d_in[3];   // (128,128)
  const float* cache_k  = (const float*)d_in[4];   // (32,8,4096,128)
  const float* cache_v  = (const float*)d_in[5];   // (32,8,4096,128)
  const float* mask     = (const float*)d_in[6];   // (1,32,32,4096)
  const int*   pos_p    = (const int*)d_in[7];     // scalar

  float* ws        = (float*)d_ws;
  float* xqkv      = ws;                               // 32*6144   = 196608
  float* q_rot     = xqkv     + 32 * 6144;             // 1024*128  = 131072
  float* k_rot     = q_rot    + 1024 * HD;             // 256*128   = 32768
  float* v_new     = k_rot    + 256 * HD;              // 256*128   = 32768
  float* partials  = v_new    + 256 * HD;              // 256*16*520 = 2129920
  float* attn_out  = partials + 256 * NPART * PSTRIDE; // 131072
  float* gemm_part = attn_out + 32 * DIM;              // 4*196608  = 786432
  // total ~3.44M floats (~13.8 MB) of workspace

  // 1) QKV projection: (32x4096) @ (4096x6144), split-K=4 + reduce
  wmma_gemm32_splitk<<<dim3(6144 / 128, NSPLITK), 256, 0, stream>>>(
      x, wqkv, gemm_part, DIM, 6144);
  reduce_splitk<<<(32 * 6144) / 256, 256, 0, stream>>>(gemm_part, xqkv, 32 * 6144);
  // 2) RoPE for Q and K + V extraction
  rope_kernel<<<80, 256, 0, stream>>>(xqkv, rot_mat, q_rot, k_rot, v_new);
  // 3) Flash-decode attention, 1024 workgroups
  attn_kernel<<<BATCH * NKV * NSPLIT, 128, 0, stream>>>(
      q_rot, k_rot, v_new, cache_k, cache_v, mask, pos_p, partials);
  // 4) Combine split partials
  combine_kernel<<<BATCH * NKV, HD, 0, stream>>>(partials, attn_out);
  // 5) Output projection: (32x4096) @ (4096x4096), split-K=4 + reduce
  wmma_gemm32_splitk<<<dim3(DIM / 128, NSPLITK), 256, 0, stream>>>(
      attn_out, wo, gemm_part, DIM, DIM);
  reduce_splitk<<<(32 * DIM) / 256, 256, 0, stream>>>(gemm_part, (float*)d_out, 32 * DIM);
}
